// ABNet_sc_51024211476541
// MI455X (gfx1250) — matrix operand order repaired and checked by reading, not compiled, not run on hardware
//
#include <hip/hip_runtime.h>
#include <hip/hip_bf16.h>
#include <math.h>

// ---------------------------------------------------------------------------
// ABNet fused ensemble MLP + CBF-QP filter for MI455X (gfx1250, wave32, WMMA)
//
// ~1.37 TFLOP dominated by two batched (10x) 32768x1024x1024 GEMMs.
// Weights (42MB bf16 packed) are L2-resident (192MB L2); h1 lives in LDS
// (258KB/WGP tile); the 1024-wide x21/x22 intermediates are folded into the
// 2-channel W31/W32 projections in registers. HBM traffic ~90MB -> compute
// bound -> v_wmma_f32_16x16x32_bf16 with fp32 accumulate.
//
// This revision: the 256-VGPR A cache (which starved the allocator and
// forced s_wait_loadcnt 0 before every WMMA) is replaced by a 4-stage
// rotating register pipeline over A (from LDS, DScnt) and B21/B22 (from
// global, LOADcnt): stage kt+3 loads issue before stage kt is consumed, so
// 3 stages of loads are always in flight behind the matrix pipe. An opaque
// zero offset (asm "+v") keeps the compiler from re-hoisting the
// nt-invariant A loads back into a register cache.
// ---------------------------------------------------------------------------

typedef __attribute__((ext_vector_type(16))) __bf16 v16bf;
typedef __attribute__((ext_vector_type(8)))  __bf16 v8bf;
typedef __attribute__((ext_vector_type(8)))  float  v8f;

#define H_     10
#define B_     32768
#define FIN    4
#define H1_    1024
#define NT_    64          // n-tiles of 16 over 1024
#define KT_    32          // k-tiles of 32 over 1024
#define MTILE  128
#define TPB    256
#define PIPE   4           // software pipeline depth (stages in flight: 3)
#define PITCH  (H1_ + 8)   // bf16 elems per LDS row (pad 16B -> stride 4 banks)

#define PACK_HEAD_ELEMS (H1_ * H1_)            // bf16 per head per matrix
#define PACK_MAT_ELEMS  (H_ * PACK_HEAD_ELEMS) // bf16 per matrix
// workspace byte offsets
#define WS_PACK21  ((size_t)0)
#define WS_PACK22  (WS_PACK21 + (size_t)PACK_MAT_ELEMS * 2)
#define WS_X31     (WS_PACK22 + (size_t)PACK_MAT_ELEMS * 2)
#define WS_X32     (WS_X31 + (size_t)H_ * B_ * 2 * 4)
#define SMEM_BYTES (2048 + MTILE * PITCH * 2)  // x-tile staging + h1 tile

// ---------------------------------------------------------------------------
// Kernel 0: pack W21/W22 fp32 -> bf16 in the wave32 WMMA B-matrix layout.
// B (32x16 bf16): lane l -> n = l%16, k = 8*(l/16) + e%8 + 16*(e/8).  Block(nt,kt) =
// 512 bf16 stored as [half(2)][lane(32)][8 bf16] -> two coalesced 16B
// vectors per lane in the GEMM.
// ---------------------------------------------------------------------------
__global__ void __launch_bounds__(TPB) pack_weights(
    const float* __restrict__ W21, const float* __restrict__ W22,
    __bf16* __restrict__ out) {
  unsigned v = blockIdx.x * TPB + threadIdx.x;   // one 8-elem (16B) vector
  unsigned lane = v & 31u;  v >>= 5;
  unsigned half = v & 1u;   v >>= 1;
  unsigned kt   = v & 31u;  v >>= 5;
  unsigned nt   = v & 63u;  v >>= 6;
  unsigned h    = v % H_;
  unsigned mat  = v / H_;                        // 0 = W21, 1 = W22
  const float* src = (mat ? W22 : W21) + (size_t)h * (H1_ * H1_);
  unsigned n     = nt * 16 + (lane & 15u);
  unsigned kbase = kt * 32 + (lane >> 4) * 8 + half * 16;
  __bf16* dst = out + (size_t)mat * PACK_MAT_ELEMS + (size_t)h * PACK_HEAD_ELEMS
              + ((size_t)(nt * KT_ + kt)) * 512 + half * 256 + lane * 8;
#pragma unroll
  for (int j = 0; j < 8; ++j)
    dst[j] = (__bf16)src[(size_t)(kbase + j) * H1_ + n];
}

// ---- fragment loaders -----------------------------------------------------
__device__ __forceinline__ v16bf loadB(const __bf16* p) {
  v8bf lo = *(const v8bf*)p;          // elems 0..7  (k = 8g + 0..7)
  v8bf hi = *(const v8bf*)(p + 256);  // elems 8..15 (k = 16 + 8g + 0..7)
  return __builtin_shufflevector(lo, hi, 0,1,2,3,4,5,6,7,8,9,10,11,12,13,14,15);
}
__device__ __forceinline__ v16bf loadA(const __bf16* p) {
  v8bf lo = *(const v8bf*)p;          // k = K0 + 8g + 0..7
  v8bf hi = *(const v8bf*)(p + 16);   // k = K0 + 16 + 8g + 0..7
  return __builtin_shufflevector(lo, hi, 0,1,2,3,4,5,6,7,8,9,10,11,12,13,14,15);
}

// ---------------------------------------------------------------------------
// Kernel 1: per (head, 128-row tile): h1 -> LDS(bf16); dual fused GEMM via
// WMMA with a 4-stage register pipeline; per-n-tile bias+ReLU+projection
// fold; write x31/x32 (H,B,2) f32 to workspace.
// ---------------------------------------------------------------------------
__global__ void __launch_bounds__(TPB) abnet_main(
    const float* __restrict__ x,   const float* __restrict__ W1,
    const float* __restrict__ b1,  const float* __restrict__ b21,
    const float* __restrict__ b22, const float* __restrict__ W31,
    const float* __restrict__ b31, const float* __restrict__ W32,
    const float* __restrict__ b32,
    const __bf16* __restrict__ pack21, const __bf16* __restrict__ pack22,
    float* __restrict__ x31ws, float* __restrict__ x32ws) {
  extern __shared__ char smem[];
  float*  xs = (float*)smem;                 // [128][4] staged input rows
  __bf16* h1 = (__bf16*)(smem + 2048);       // [128][PITCH] bf16 activations

  const int h    = blockIdx.y;
  const int row0 = blockIdx.x * MTILE;
  const int t    = threadIdx.x;

  // ---- stage x tile ----
  if (t < MTILE) ((float4*)xs)[t] = ((const float4*)x)[row0 + t];
  __syncthreads();

  // ---- layer 1: h1 = relu(x @ W1[h] + b1[h]), K=4 -> VALU FMAs ----
  {
    float w1c[FIN][4], b1c[4];
#pragma unroll
    for (int j = 0; j < 4; ++j) {
      int o = t + j * TPB;
      b1c[j] = b1[h * H1_ + o];
#pragma unroll
      for (int f = 0; f < FIN; ++f) w1c[f][j] = W1[(h * FIN + f) * H1_ + o];
    }
    for (int m = 0; m < MTILE; ++m) {
      float4 xr = ((const float4*)xs)[m];
#pragma unroll
      for (int j = 0; j < 4; ++j) {
        float s = b1c[j] + xr.x * w1c[0][j] + xr.y * w1c[1][j]
                         + xr.z * w1c[2][j] + xr.w * w1c[3][j];
        h1[m * PITCH + t + j * TPB] = (__bf16)fmaxf(s, 0.0f);
      }
    }
  }
  __syncthreads();

  // ---- fused GEMMs: each wave owns a 16-row M strip ----
  const int lane = t & 31;
  const int wv   = t >> 5;        // wave id 0..7
  const int g    = lane >> 4;     // half-wave group
  const int nl   = lane & 15;     // n within tile / A-row selector
  // A (16x32 bf16): lane l -> m = l%16; elem e -> k = (e>=8?16:0) + e%8 + 8g
  const __bf16* aRow = h1 + (wv * 16 + nl) * PITCH + 8 * g;
  const __bf16* bp21 = pack21 + (size_t)h * PACK_HEAD_ELEMS;
  const __bf16* bp22 = pack22 + (size_t)h * PACK_HEAD_ELEMS;

  float x31p[8][2] = {}, x32p[8][2] = {};

  // opaque zero: makes the A LDS address formally nt-variant so the
  // compiler cannot LICM the A loads into a 256-VGPR cache (round-1 pathology)
  unsigned aoff = 0;

  for (int nt = 0; nt < NT_; ++nt) {
    const __bf16* blk21 = bp21 + (size_t)nt * KT_ * 512 + lane * 8;
    const __bf16* blk22 = bp22 + (size_t)nt * KT_ * 512 + lane * 8;
    // pull next n-tile's weight blocks toward L1 while crunching this one
    __builtin_prefetch(blk21 + (size_t)KT_ * 512, 0, 1);
    __builtin_prefetch(blk22 + (size_t)KT_ * 512, 0, 1);

    v8f acc21 = {}, acc22 = {};

    // ---- 4-stage rotating pipeline: prime 3 stages ----
    v16bf ab[PIPE], b1b[PIPE], b2b[PIPE];
#pragma unroll
    for (int s = 0; s < PIPE - 1; ++s) {
      ab[s]  = loadA(aRow + aoff + s * 32);
      b1b[s] = loadB(blk21 + (size_t)s * 512);
      b2b[s] = loadB(blk22 + (size_t)s * 512);
    }
#pragma unroll
    for (int kt = 0; kt < KT_; ++kt) {
      const int cur = kt % PIPE;
      const int pk  = kt + PIPE - 1;        // stage to fetch ahead
      if (pk < KT_) {
        const int nxt = pk % PIPE;
        ab[nxt]  = loadA(aRow + aoff + pk * 32);
        b1b[nxt] = loadB(blk21 + (size_t)pk * 512);
        b2b[nxt] = loadB(blk22 + (size_t)pk * 512);
      }
      acc21 = __builtin_amdgcn_wmma_f32_16x16x32_bf16(false, ab[cur], false, b1b[cur],
                                                      (short)0, acc21, false, false);
      acc22 = __builtin_amdgcn_wmma_f32_16x16x32_bf16(false, ab[cur], false, b2b[cur],
                                                      (short)0, acc22, false, false);
    }

    // fold this 16-wide n-slab into the 2-channel projections
    const int nglob = nt * 16 + nl;
    const float bi21 = b21[h * H1_ + nglob];
    const float bi22 = b22[h * H1_ + nglob];
    const float w31c0 = W31[(h * H1_ + nglob) * 2 + 0];
    const float w31c1 = W31[(h * H1_ + nglob) * 2 + 1];
    const float w32c0 = W32[(h * H1_ + nglob) * 2 + 0];
    const float w32c1 = W32[(h * H1_ + nglob) * 2 + 1];
#pragma unroll
    for (int r = 0; r < 8; ++r) {
      float v21 = fmaxf(acc21[r] + bi21, 0.0f);   // relu(x21)
      x31p[r][0] += v21 * w31c0;  x31p[r][1] += v21 * w31c1;
      float v22 = fmaxf(acc22[r] + bi22, 0.0f);   // relu(x22)
      x32p[r][0] += v22 * w32c0;  x32p[r][1] += v22 * w32c1;
    }

    asm volatile("" : "+v"(aoff));   // keep A loads inside the nt loop
  }

  // ---- reduce over the 16 n-lanes of each half-wave, write results ----
#pragma unroll
  for (int r = 0; r < 8; ++r) {
#pragma unroll
    for (int c = 0; c < 2; ++c) {
      float a31 = x31p[r][c], a32 = x32p[r][c];
#pragma unroll
      for (int off = 1; off < 16; off <<= 1) {   // xor<16 stays in half-wave
        a31 += __shfl_xor(a31, off, 32);
        a32 += __shfl_xor(a32, off, 32);
      }
      if (nl == 0) {
        int row = row0 + wv * 16 + r + 8 * g;    // C/D: m = r + 8*(lane/16)
        x31ws[((size_t)h * B_ + row) * 2 + c] = a31 + b31[h * 2 + c];
        float tt = a32 + b32[h * 2 + c];
        x32ws[((size_t)h * B_ + row) * 2 + c] = 4.0f / (1.0f + __expf(-tt));
      }
    }
  }
}

// ---------------------------------------------------------------------------
// Kernel 2: per-row physics + QP filter + softmax-weighted head sum.
// (p1 = head 0's x32[:,0] feeds every head's h_qp -> needs all heads done.)
// ---------------------------------------------------------------------------
__global__ void __launch_bounds__(TPB) abnet_epilogue(
    const float* __restrict__ x,
    const float* __restrict__ x31ws, const float* __restrict__ x32ws,
    const float* __restrict__ wt,
    const float* __restrict__ meanp, const float* __restrict__ stdp,
    float* __restrict__ out) {
  const int b = blockIdx.x * TPB + threadIdx.x;
  float4 xr = ((const float4*)x)[b];
  float px = xr.x * stdp[0] + meanp[0];
  float py = xr.y * stdp[1] + meanp[1];
  float th = xr.z * stdp[2] + meanp[2];
  float v  = xr.w * stdp[3] + meanp[3];
  float st = __sinf(th), ct = __cosf(th);
  float dx = px - 40.0f, dy = py - 15.0f;
  float barrier = dx * dx + dy * dy - 36.0f;
  float bdot    = 2.0f * dx * v * ct + 2.0f * dy * v * st;
  float Lf2b    = 2.0f * v * v;
  float G0 = -(-2.0f * dx * v * st + 2.0f * dy * v * ct);  // -LgLfbu1
  float G1 = -( 2.0f * dx * ct + 2.0f * dy * st);          // -LgLfbu2
  float GG = G0 * G0 + G1 * G1;

  // softmax over the 10 head weights (tiny, recomputed per thread)
  float wl[H_], wmax = -1e30f;
#pragma unroll
  for (int h = 0; h < H_; ++h) { wl[h] = wt[h]; wmax = fmaxf(wmax, wl[h]); }
  float wsum = 0.0f;
#pragma unroll
  for (int h = 0; h < H_; ++h) { wl[h] = __expf(wl[h] - wmax); wsum += wl[h]; }
  float winv = 1.0f / wsum;

  const float p1 = x32ws[(size_t)b * 2 + 0];       // head 0, channel 0
  float acc0 = 0.0f, acc1 = 0.0f;
#pragma unroll
  for (int h = 0; h < H_; ++h) {
    size_t base = ((size_t)h * B_ + b) * 2;
    float z0 = -x31ws[base + 0];
    float z1 = -x31ws[base + 1];
    float a  = (h == 0) ? x32ws[(size_t)b * 2 + 1] : x32ws[base + 0];
    float hqp = Lf2b + (p1 + a) * bdot + p1 * a * barrier;
    float gz  = G0 * z0 + G1 * z1;
    float lam = fmaxf(gz - hqp, 0.0f) / (GG + 1e-12f);
    float u0 = z0 - lam * G0;
    float u1 = z1 - lam * G1;
    float w = wl[h] * winv;
    acc0 += w * u0;
    acc1 += w * u1;
  }
  out[(size_t)b * 2 + 0] = acc0;
  out[(size_t)b * 2 + 1] = acc1;
}

// ---------------------------------------------------------------------------
extern "C" void kernel_launch(void* const* d_in, const int* in_sizes, int n_in,
                              void* d_out, int out_size, void* d_ws, size_t ws_size,
                              hipStream_t stream) {
  const float* x    = (const float*)d_in[0];
  const float* W1   = (const float*)d_in[1];
  const float* b1   = (const float*)d_in[2];
  const float* W21  = (const float*)d_in[3];
  const float* b21  = (const float*)d_in[4];
  const float* W22  = (const float*)d_in[5];
  const float* b22  = (const float*)d_in[6];
  const float* W31  = (const float*)d_in[7];
  const float* b31  = (const float*)d_in[8];
  const float* W32  = (const float*)d_in[9];
  const float* b32  = (const float*)d_in[10];
  const float* wt   = (const float*)d_in[11];
  const float* mean = (const float*)d_in[12];
  const float* stdv = (const float*)d_in[13];
  float* out = (float*)d_out;

  char* ws = (char*)d_ws;
  __bf16* pack   = (__bf16*)(ws + WS_PACK21);
  float*  x31ws  = (float*)(ws + WS_X31);
  float*  x32ws  = (float*)(ws + WS_X32);

  // 2 mats * 10 heads * 64nt * 32kt * 2half * 32lane 16B-vectors / 256 threads
  pack_weights<<<10240, TPB, 0, stream>>>(W21, W22, pack);

  dim3 grid(B_ / MTILE, H_);
  abnet_main<<<grid, TPB, SMEM_BYTES, stream>>>(
      x, W1, b1, b21, b22, W31, b31, W32, b32,
      pack, pack + (size_t)PACK_MAT_ELEMS, x31ws, x32ws);

  abnet_epilogue<<<B_ / TPB, TPB, 0, stream>>>(x, x31ws, x32ws, wt, mean, stdv, out);
}
